// HierarchicalExpertGating_37864431681651
// MI455X (gfx1250) — matrix operand order repaired and checked by reading, not compiled, not run on hardware
//
#include <hip/hip_runtime.h>
#include <math.h>

// Problem constants (from reference): B=4,S=4096 -> N=16384 tokens, H=2048,
// G=4 groups x 16 experts = 64 experts, conf hidden = H/4 = 512.
#define H     2048
#define NTOK  16384
#define C1    512
#define EPG   16
#define NG    4
#define NE    64
#define BCOLS 80   // 64 expert cols + 4 group cols + 12 zero pad
#define KCH   128  // TDM K-chunk staged into LDS per step
#define NCHUNK (H / KCH)

typedef __attribute__((ext_vector_type(16))) __bf16       v16bf;
typedef __attribute__((ext_vector_type(8)))  float        v8f;
typedef __attribute__((ext_vector_type(4)))  float        v4f;
typedef __attribute__((ext_vector_type(4)))  unsigned int v4u;
typedef __attribute__((ext_vector_type(8)))  int          v8i;
typedef __attribute__((ext_vector_type(4)))  int          v4i;

union BF16x16 { v16bf v; unsigned short u[16]; v4u q[2]; };

__device__ __forceinline__ unsigned short f2bf(float f) {
  union { float f; unsigned int u; } t; t.f = f;
  unsigned int r = t.u + 0x7FFFu + ((t.u >> 16) & 1u);  // round-to-nearest-even
  return (unsigned short)(r >> 16);
}

__device__ __forceinline__ v8f wmma_bf16(v16bf a, v16bf b, v8f c) {
  // D(16x16 f32) = A(16x32 bf16) x B(32x16 bf16) + C
  return __builtin_amdgcn_wmma_f32_16x16x32_bf16(false, a, false, b, (short)0, c,
                                                 false, false);
}

// TDM: stream a [BCOLS rows x KCH cols] bf16 tile of bcat (row stride H elems)
// into LDS at lds_addr, packed contiguously as [BCOLS][KCH].
// D# group0: count=1 | lds_addr | global_addr[56:0] | type=2.
// D# group1: data_size=1(2B), tensor_dim0=H, tensor_dim1=BCOLS,
//            tile_dim0=KCH, tile_dim1=BCOLS, tensor_dim0_stride=H.
// This toolchain's builtin takes 6 args: (g0, g1, g2, g3, g_ext, cpol).
__device__ __forceinline__ void tdm_load_bcat_tile(unsigned lds_addr,
                                                   const unsigned short* gptr) {
  unsigned long long ga = (unsigned long long)(uintptr_t)gptr;
  v4u g0;
  g0[0] = 1u;                                   // count=1, is_restore=0
  g0[1] = lds_addr;                             // lds_addr[31:0]
  g0[2] = (unsigned)ga;                         // global_addr[31:0]
  g0[3] = ((unsigned)(ga >> 32) & 0x01FFFFFFu)  // global_addr[56:32]
          | 0x80000000u;                        // type=2 ("image") in bits 127:126
  v8i g1;
  g1[0] = 0x00010000;                                   // data_size=1 (2 bytes)
  g1[1] = (int)(((unsigned)H & 0xFFFFu) << 16);         // tensor_dim0[15:0]
  g1[2] = (int)(((unsigned)H >> 16) |
                (((unsigned)BCOLS & 0xFFFFu) << 16));   // dim0 hi | tensor_dim1 lo
  g1[3] = (int)(((unsigned)BCOLS >> 16) |
                ((unsigned)KCH << 16));                 // dim1 hi | tile_dim0
  g1[4] = BCOLS;                                        // tile_dim1 (tile_dim2=0)
  g1[5] = H;                                            // tensor_dim0_stride[31:0]
  g1[6] = 0;                                            // stride hi | dim1_stride lo
  g1[7] = 0;
  v4i z4; z4[0] = 0; z4[1] = 0; z4[2] = 0; z4[3] = 0;   // 2-D: groups 2/3 unused
  v8i z8; z8[0] = 0; z8[1] = 0; z8[2] = 0; z8[3] = 0;
  z8[4] = 0; z8[5] = 0; z8[6] = 0; z8[7] = 0;
  __builtin_amdgcn_tensor_load_to_lds(g0, g1, z4, z4, z8, 0);
}

// ---------------------------------------------------------------------------
// Prep: Wc1 (H x C1 fp32) -> transposed bf16 [C1][H] so each lane's B fragment
// (16 consecutive K for a fixed column) is one contiguous 32B load.
// ---------------------------------------------------------------------------
__global__ __launch_bounds__(256) void prep_wc1t(const float* __restrict__ wc1,
                                                 unsigned short* __restrict__ w1t) {
  int i = blockIdx.x * 256 + threadIdx.x;      // [0, C1*H)
  int k = i & (H - 1);
  int c = i >> 11;
  w1t[i] = f2bf(wc1[(size_t)k * C1 + c]);
}

// Prep: concatenated transposed bf16 B [BCOLS][H]:
//   cols 0..63  : We[g][k][e] with c = g*16+e
//   cols 64..67 : Wg[k][c-64]
//   cols 68..79 : 0
__global__ __launch_bounds__(256) void prep_bcat(const float* __restrict__ We,
                                                 const float* __restrict__ Wg,
                                                 unsigned short* __restrict__ bcat) {
  int i = blockIdx.x * 256 + threadIdx.x;      // [0, BCOLS*H)
  int k = i & (H - 1);
  int c = i >> 11;
  float v = 0.0f;
  if (c < NE) {
    int g = c >> 4, e = c & 15;
    v = We[(((size_t)g * H) + (size_t)k) * EPG + e];
  } else if (c < NE + NG) {
    v = Wg[(size_t)k * NG + (c - NE)];
  }
  bcat[i] = f2bf(v);
}

// ---------------------------------------------------------------------------
// conf[n] = sigmoid( gelu(x @ Wc1 + bc1) @ Wc2 + bc2 )
// Block: 16 rows x all 512 cols. 8 waves x 4 col-tiles of 16. A tile staged
// through LDS (shared by all 8 waves). Deterministic reduction (shuffle tree
// + per-wave LDS slots, fixed summation order) -> bitwise stable per replay.
// ---------------------------------------------------------------------------
__global__ __launch_bounds__(256) void conf_kernel(const float* __restrict__ x,
                                                   const unsigned short* __restrict__ w1t,
                                                   const float* __restrict__ bc1,
                                                   const float* __restrict__ wc2,
                                                   const float* __restrict__ bc2,
                                                   float* __restrict__ conf) {
  __shared__ __align__(16) unsigned short aT[16 * 32];  // A tile, bf16 row-major
  __shared__ float wsum[8][16];
  const int tid  = threadIdx.x;
  const int wave = tid >> 5;
  const int lane = tid & 31;
  const int half = lane >> 4;
  const int l16  = lane & 15;
  const int row0 = blockIdx.x * 16;

  v8f acc[4] = {};
  for (int ks = 0; ks < H; ks += 32) {
    // Stage 16x32 fp32 -> bf16 into LDS cooperatively.
    #pragma unroll
    for (int i = 0; i < 2; ++i) {
      int idx = tid + i * 256;
      int r = idx >> 5, k = idx & 31;
      aT[idx] = f2bf(x[(size_t)(row0 + r) * H + ks + k]);
    }
    __syncthreads();
    // A fragment: lane(row=l16): K = half*16 + 0..15 contiguous
    BF16x16 a;
    a.q[0] = *(const v4u*)&aT[l16 * 32 + half * 16];
    a.q[1] = *(const v4u*)&aT[l16 * 32 + half * 16 + 8];
    #pragma unroll
    for (int t = 0; t < 4; ++t) {
      int col = (wave * 4 + t) * 16 + l16;
      // B fragment: lane(col): K = half*16 + 0..15 contiguous (transposed store)
      v16bf b = *(const v16bf*)(w1t + (size_t)col * H + ks + half * 16);
      acc[t] = wmma_bf16(a.v, b, acc[t]);
    }
    __syncthreads();  // protect aT before next staging
  }

  // Epilogue: gelu(h+bc1)*wc2, reduce over columns per row.
  float psum[8] = {0, 0, 0, 0, 0, 0, 0, 0};
  #pragma unroll
  for (int t = 0; t < 4; ++t) {
    int c = (wave * 4 + t) * 16 + l16;
    float b1 = bc1[c], w2 = wc2[c];
    #pragma unroll
    for (int j = 0; j < 8; ++j) {
      float v = acc[t][j] + b1;                           // C layout: M=j+8*half, N=l16
      float g = 0.5f * v * (1.0f + erff(v * 0.70710678118654752f));
      psum[j] += g * w2;
    }
  }
  #pragma unroll
  for (int j = 0; j < 8; ++j) {
    float s = psum[j];
    s += __shfl_xor(s, 1);
    s += __shfl_xor(s, 2);
    s += __shfl_xor(s, 4);
    s += __shfl_xor(s, 8);          // sum over the 16 columns in this half
    if (l16 == 0) wsum[wave][j + half * 8] = s;
  }
  __syncthreads();
  if (tid < 16) {
    float s = 0.0f;
    #pragma unroll
    for (int w = 0; w < 8; ++w) s += wsum[w][tid];        // fixed order: deterministic
    s += bc2[0];
    conf[row0 + tid] = 1.0f / (1.0f + expf(-s));
  }
}

// ---------------------------------------------------------------------------
// logits = x @ [We | Wg | 0]  (WMMA) with the B panel streamed into LDS by the
// Tensor Data Mover (double-buffered, TENSORcnt-tracked), then the exact
// reference gating chain. Block: 64 rows, 4 waves x 16 rows x 5 col-tiles.
// ---------------------------------------------------------------------------
__global__ __launch_bounds__(128) void gating_kernel(const float* __restrict__ x,
                                                     const unsigned short* __restrict__ bcat,
                                                     const float* __restrict__ bg,
                                                     const float* __restrict__ be,
                                                     const float* __restrict__ prio,
                                                     const float* __restrict__ conf,
                                                     float* __restrict__ out) {
  __shared__ __align__(32) unsigned short bB[2][BCOLS * KCH];  // TDM tiles, 2x20KB
  __shared__ float lg[4][16][BCOLS];   // per-wave 16x80 logit tile
  __shared__ float sc[64][NE];         // per-token expert scores
  const int tid  = threadIdx.x;
  const int wave = tid >> 5;
  const int lane = tid & 31;
  const int half = lane >> 4;
  const int l16  = lane & 15;
  const long long rowA = (long long)blockIdx.x * 64 + wave * 16 + l16;

  const unsigned ldsB0 = (unsigned)(uintptr_t)&bB[0][0];
  const unsigned ldsB1 = (unsigned)(uintptr_t)&bB[1][0];

  // Prologue: wave 0 launches the first two B-panel tiles.
  if (wave == 0) {
    tdm_load_bcat_tile(ldsB0, bcat + 0 * KCH);
    tdm_load_bcat_tile(ldsB1, bcat + 1 * KCH);
  }

  v8f acc[5] = {};
  const float* xr = x + (size_t)rowA * H;
  for (int c = 0; c < NCHUNK; ++c) {
    if (wave == 0) {
      if (c < NCHUNK - 1) __builtin_amdgcn_s_wait_tensorcnt(1);  // oldest tile done
      else                __builtin_amdgcn_s_wait_tensorcnt(0);
    }
    __syncthreads();                       // tile c visible to all waves
    const unsigned short* Bs = &bB[c & 1][0];
    #pragma unroll
    for (int ksub = 0; ksub < KCH; ksub += 32) {
      const int ks = c * KCH + ksub;
      BF16x16 a;
      v4f f0 = *(const v4f*)(xr + ks + half * 16);
      v4f f1 = *(const v4f*)(xr + ks + half * 16 + 4);
      v4f f2 = *(const v4f*)(xr + ks + half * 16 + 8);
      v4f f3 = *(const v4f*)(xr + ks + half * 16 + 12);
      a.u[0]  = f2bf(f0.x); a.u[1]  = f2bf(f0.y); a.u[2]  = f2bf(f0.z); a.u[3]  = f2bf(f0.w);
      a.u[4]  = f2bf(f1.x); a.u[5]  = f2bf(f1.y); a.u[6]  = f2bf(f1.z); a.u[7]  = f2bf(f1.w);
      a.u[8]  = f2bf(f2.x); a.u[9]  = f2bf(f2.y); a.u[10] = f2bf(f2.z); a.u[11] = f2bf(f2.w);
      a.u[12] = f2bf(f3.x); a.u[13] = f2bf(f3.y); a.u[14] = f2bf(f3.z); a.u[15] = f2bf(f3.w);
      #pragma unroll
      for (int t = 0; t < 5; ++t) {
        // B fragment from staged tile: col-major rows of KCH contiguous bf16.
        v16bf b = *(const v16bf*)(Bs + (t * 16 + l16) * KCH + ksub + half * 16);
        acc[t] = wmma_bf16(a.v, b, acc[t]);
      }
    }
    __syncthreads();                       // all waves done reading buf[c&1]
    if (wave == 0 && c + 2 < NCHUNK)       // refill the buffer just consumed
      tdm_load_bcat_tile((c & 1) ? ldsB1 : ldsB0,
                         bcat + (size_t)(c + 2) * KCH);
  }

  // Scatter C fragments: VGPR j -> row (j + 8*half), col (t*16 + l16).
  #pragma unroll
  for (int t = 0; t < 5; ++t)
    #pragma unroll
    for (int j = 0; j < 8; ++j)
      lg[wave][j + half * 8][t * 16 + l16] = acc[t][j];
  __syncthreads();

  if (tid < 64) {
    const int wv = tid >> 4, rr = tid & 15;
    const long long row = (long long)blockIdx.x * 64 + tid;
    const float* L = &lg[wv][rr][0];
    const float cnf = conf[row];

    // Group softmax + top-2 groups (lax.top_k tie-break: lowest index wins).
    float gl[NG];
    #pragma unroll
    for (int g = 0; g < NG; ++g) gl[g] = L[NE + g] + bg[g];
    float gm = fmaxf(fmaxf(gl[0], gl[1]), fmaxf(gl[2], gl[3]));
    float gp[NG], gs = 0.0f;
    #pragma unroll
    for (int g = 0; g < NG; ++g) { gp[g] = expf(gl[g] - gm); gs += gp[g]; }
    float ginv = 1.0f / gs;
    #pragma unroll
    for (int g = 0; g < NG; ++g) gp[g] *= ginv;
    int g0 = 0; float gv0 = gp[0];
    #pragma unroll
    for (int g = 1; g < NG; ++g) if (gp[g] > gv0) { gv0 = gp[g]; g0 = g; }
    int g1 = -1; float gv1 = -1e30f;
    #pragma unroll
    for (int g = 0; g < NG; ++g) if (g != g0 && gp[g] > gv1) { gv1 = gp[g]; g1 = g; }

    float* S = sc[tid];
    #pragma unroll
    for (int i = 0; i < NE; ++i) S[i] = 0.0f;

    const int   gsel[2] = {g0, g1};
    const float gwt[2]  = {gv0, gv1};
    for (int s = 0; s < 2; ++s) {
      const int g = gsel[s];
      float p[EPG], m = -1e30f;
      #pragma unroll
      for (int e = 0; e < EPG; ++e) {
        float le = (L[g * EPG + e] + be[g * EPG + e]) * prio[g * EPG + e];
        p[e] = le; m = fmaxf(m, le);
      }
      float sum = 0.0f;
      #pragma unroll
      for (int e = 0; e < EPG; ++e) { p[e] = expf(p[e] - m); sum += p[e]; }
      const float pin = cnf / sum;
      const float mix = (1.0f - cnf) * (1.0f / 16.0f);
      #pragma unroll
      for (int e = 0; e < EPG; ++e) p[e] = p[e] * pin + mix;
      int e0 = 0; float pv0 = p[0];
      #pragma unroll
      for (int e = 1; e < EPG; ++e) if (p[e] > pv0) { pv0 = p[e]; e0 = e; }
      int e1 = -1; float pv1 = -1e30f;
      #pragma unroll
      for (int e = 0; e < EPG; ++e) if (e != e0 && p[e] > pv1) { pv1 = p[e]; e1 = e; }
      S[g * EPG + e0] = pv0 * gwt[s];
      S[g * EPG + e1] = pv1 * gwt[s];
    }

    float tot = 0.0f;
    #pragma unroll
    for (int i = 0; i < NE; ++i) tot += S[i];
    const float inv = 1.0f / (tot + 1e-9f);
    int i0 = 0; float sv0 = S[0];
    #pragma unroll
    for (int i = 1; i < NE; ++i) { float v = S[i]; if (v > sv0) { sv0 = v; i0 = i; } }
    int i1 = -1; float sv1 = -1e30f;
    #pragma unroll
    for (int i = 0; i < NE; ++i) { float v = S[i]; if (i != i0 && v > sv1) { sv1 = v; i1 = i; } }
    const float d0 = sv0 * inv, d1 = sv1 * inv;

    const size_t o1 = (size_t)row * NE;                  // dispatch
    const size_t o2 = (size_t)NTOK * NE + o1;            // combine (== dispatch)
    #pragma unroll
    for (int i = 0; i < NE; ++i) {
      float v = (i == i0) ? d0 : ((i == i1) ? d1 : 0.0f);
      out[o1 + i] = v;
      out[o2 + i] = v;
    }
  }
  if (blockIdx.x == 0 && tid == 0)
    out[(size_t)2 * NTOK * NE] = 0.0f;                   // aux_loss
}

// ---------------------------------------------------------------------------
// Inputs (setup_inputs order): 0:x 1:Wg 2:bg 3:We 4:be 5:Wc1 6:bc1 7:Wc2
//                              8:bc2 9:priority
// Output: dispatch (N*64) | combine (N*64) | aux_loss (1), all fp32.
// ---------------------------------------------------------------------------
extern "C" void kernel_launch(void* const* d_in, const int* in_sizes, int n_in,
                              void* d_out, int out_size, void* d_ws, size_t ws_size,
                              hipStream_t stream) {
  (void)in_sizes; (void)n_in; (void)out_size; (void)ws_size;
  const float* x    = (const float*)d_in[0];
  const float* Wg   = (const float*)d_in[1];
  const float* bg   = (const float*)d_in[2];
  const float* We   = (const float*)d_in[3];
  const float* be   = (const float*)d_in[4];
  const float* Wc1  = (const float*)d_in[5];
  const float* bc1  = (const float*)d_in[6];
  const float* Wc2  = (const float*)d_in[7];
  const float* bc2  = (const float*)d_in[8];
  const float* prio = (const float*)d_in[9];
  float* out = (float*)d_out;

  // Workspace layout (all offsets 32B-aligned):
  //   [0, 64KB)            conf  : float[NTOK]
  //   [64KB, 64KB+2MB)     w1t   : bf16[C1][H]
  //   [+2MB, +2MB+320KB)   bcat  : bf16[BCOLS][H]
  char* ws = (char*)d_ws;
  float*          conf = (float*)ws;
  unsigned short* w1t  = (unsigned short*)(ws + 65536);
  unsigned short* bcat = (unsigned short*)(ws + 65536 + (size_t)C1 * H * 2);

  prep_wc1t<<<(C1 * H) / 256, 256, 0, stream>>>(Wc1, w1t);
  prep_bcat<<<(BCOLS * H) / 256, 256, 0, stream>>>(We, Wg, bcat);
  conf_kernel<<<NTOK / 16, 256, 0, stream>>>(x, w1t, bc1, Wc2, bc2, conf);
  gating_kernel<<<NTOK / 64, 128, 0, stream>>>(x, bcat, bg, be, prio, conf, out);
}
